// MultiHeadAttention_56427280334952
// MI455X (gfx1250) — matrix operand order repaired and checked by reading, not compiled, not run on hardware
//
#include <hip/hip_runtime.h>

// ---------------- problem constants ----------------
constexpr int Bc  = 2;
constexpr int Sc  = 2048;
constexpr int Dc  = 1024;
constexpr int Hc  = 16;
constexpr int DKc = 64;       // head dim
constexpr int Mtot = Bc * Sc; // 4096 tokens
constexpr int Kdim = Dc;
constexpr int Ndim = Dc;
constexpr int LP   = 72;      // padded LDS row stride (bf16 elems): 64 + 8

// ---------------- vector types ----------------
typedef __attribute__((ext_vector_type(16))) __bf16 v16bf;
typedef __attribute__((ext_vector_type(8)))  __bf16 v8bf;
typedef __attribute__((ext_vector_type(8)))  float  v8f;
typedef __attribute__((ext_vector_type(4)))  float  v4f;
typedef __attribute__((ext_vector_type(4)))  unsigned int u32x4;
typedef __attribute__((ext_vector_type(8)))  int i32x8;
typedef __attribute__((ext_vector_type(4)))  int i32x4;

#if __has_builtin(__builtin_amdgcn_tensor_load_to_lds) && \
    __has_builtin(__builtin_amdgcn_s_wait_tensorcnt)
#define HAVE_TDM 1
#else
#define HAVE_TDM 0
#endif

// ---------------- WMMA wrapper (CDNA5 bf16, fp32 accum) ----------------
__device__ __forceinline__ v8f wmma_bf16(v16bf a, v16bf b, v8f c) {
  return __builtin_amdgcn_wmma_f32_16x16x32_bf16(
      false, a, false, b, (short)0, c, false, false);
}

// A-fragment (16x32 bf16) per ISA 7.12.2: caller passes kb = kstep + hi*8
__device__ __forceinline__ v16bf load_a_bf16(const __bf16* __restrict__ row, int kb) {
  v8bf x0 = *(const v8bf*)(row + kb);
  v8bf x1 = *(const v8bf*)(row + kb + 16);
  v16bf a;
#pragma unroll
  for (int j = 0; j < 8; ++j) { a[j] = x0[j]; a[8 + j] = x1[j]; }
  return a;
}

// B-fragment (32x16 bf16): lane holds column N=lane%16, K = 8h+j and 16+8h+j (row stride LP)
__device__ __forceinline__ v16bf load_b_bf16(const __bf16* __restrict__ p) {
  const int hb = (threadIdx.x & 31) >> 4;
  v16bf b;
#pragma unroll
  for (int j = 0; j < 8; ++j) { b[j] = p[(hb * 8 + j) * LP]; b[8 + j] = p[(16 + hb * 8 + j) * LP]; }
  return b;
}

#if HAVE_TDM
// -------- Tensor Data Mover: 2D bf16 tile -> LDS (padded 64->72 rows) -------
// D# built per CDNA5 ISA. data_size=2B. pad: 4 DWORDs
// (8 bf16) appended every 32 DWORDs (64 bf16 = one tile row).
// Toolchain carries the 6-arg builtin form:
//   (u32x4 g0, i32x8 g1, i32x4 g2, i32x4 g3, i32x8 extra, i32 cpol)
__device__ __forceinline__ void tdm_load_2d(const __bf16* lds_dst,
                                            const __bf16* gsrc,
                                            unsigned tensor_w, unsigned tensor_h,
                                            unsigned tile_w, unsigned tile_h,
                                            unsigned stride_elems, bool pad) {
  unsigned long long ga = (unsigned long long)(size_t)gsrc;
  unsigned lds_off = (unsigned)(size_t)lds_dst;   // low 32 bits = LDS byte addr
  u32x4 g0;
  g0[0] = 1u;                                     // count=1, no gather
  g0[1] = lds_off;
  g0[2] = (unsigned)(ga & 0xFFFFFFFFu);
  g0[3] = (unsigned)((ga >> 32) & 0x01FFFFFFu) | (2u << 30);  // type=2 (image)
  i32x8 g1;
  unsigned d0 = (1u << 16);                       // data_size = 2 bytes
  if (pad) d0 |= (1u << 20) | (4u << 22) | (3u << 25); // every 32 DW, add 4 DW
  g1[0] = (int)d0;
  g1[1] = (int)((tensor_w & 0xFFFFu) << 16);                       // dim0 lo16
  g1[2] = (int)(((tensor_w >> 16) & 0xFFFFu) | ((tensor_h & 0xFFFFu) << 16));
  g1[3] = (int)(((tensor_h >> 16) & 0xFFFFu) | (tile_w << 16));    // tile_dim0
  g1[4] = (int)(tile_h & 0xFFFFu);                // tile_dim1 | tile_dim2=0
  g1[5] = (int)stride_elems;                      // tensor_dim0_stride lo32
  g1[6] = 0;                                      // stride0 hi16 | stride1 lo16
  g1[7] = 0;
  i32x4 gz4 = {0, 0, 0, 0};
  i32x8 gz8 = {0, 0, 0, 0, 0, 0, 0, 0};
  __builtin_amdgcn_tensor_load_to_lds(g0, g1, gz4, gz4, gz8, 0);
}
#else
// -------- fallback: cooperative padded copy (rows x 64 tile) --------
__device__ __forceinline__ void stage_tile(__bf16* dst, const __bf16* src,
                                           int rows, int srcStride, int tid) {
  const int total = rows * 64;
  for (int e = tid * 8; e < total; e += 256 * 8) {
    const int r = e >> 6, c = e & 63;
    *(v8bf*)(dst + r * LP + c) = *(const v8bf*)(src + (size_t)r * srcStride + c);
  }
}
#endif

// ---------------- fp32 -> bf16 conversion (memory bound, one pass) ---------
__global__ __launch_bounds__(256) void cvt_kernel(const float* __restrict__ src,
                                                  __bf16* __restrict__ dst, int n) {
  const int i = (blockIdx.x * 256 + threadIdx.x) * 8;
  if (i >= n) return;
  v4f x0 = *(const v4f*)(src + i);
  v4f x1 = *(const v4f*)(src + i + 4);
  v8bf y;
#pragma unroll
  for (int j = 0; j < 4; ++j) { y[j] = (__bf16)x0[j]; y[4 + j] = (__bf16)x1[j]; }
  *(v8bf*)(dst + i) = y;
}

// ---------------- projection GEMM: Y = Xb(4096x1024) * Wb(1024x1024) + bias -
// Block: 8 waves = 128 rows x 64 cols. W 32x64 tile TDM-staged to LDS, shared
// by all waves; double buffered; each wave: 16x64 tile, 4 WMMAs / k-step.
// mode 0: fp32 out [M,N] (-> d_out)   mode 1: bf16 [B,H,S,DK]   mode 2: bf16 [B,H,DK,S]
__global__ __launch_bounds__(256) void proj_kernel(
    const __bf16* __restrict__ X, const __bf16* __restrict__ W,
    const float* __restrict__ bias, float* __restrict__ outF,
    __bf16* __restrict__ outB, int mode)
{
  __shared__ __align__(16) __bf16 Wst[2][32 * LP];

  const int tid  = threadIdx.x;
  const int lane = tid & 31;
  const int wave = tid >> 5;
  const int l15  = lane & 15;
  const int hi   = lane >> 4;

  const int nGroups = Ndim / 64;                 // 16
  const int mb = blockIdx.x / nGroups;           // 0..31 (128-row block)
  const int ng = blockIdx.x % nGroups;           // 0..15
  const int m0 = (mb * 8 + wave) * 16;
  const int n0 = ng * 64;

  const __bf16* arow = X + (size_t)(m0 + l15) * Kdim;

  v8f acc[4] = {};
  constexpr int KSTEPS = Kdim / 32;              // 32

#if HAVE_TDM
  if (wave == 0)
    tdm_load_2d(&Wst[0][0], W + n0, Ndim, Kdim, 64, 32, Ndim, true);
#endif

  for (int ki = 0; ki < KSTEPS; ++ki) {
    const int cur = ki & 1;
#if HAVE_TDM
    if (wave == 0) {
      if (ki + 1 < KSTEPS) {
        tdm_load_2d(&Wst[cur ^ 1][0], W + (size_t)(ki + 1) * 32 * Ndim + n0,
                    Ndim, Kdim, 64, 32, Ndim, true);
        __builtin_amdgcn_s_wait_tensorcnt(1);
      } else {
        __builtin_amdgcn_s_wait_tensorcnt(0);
      }
    }
    __syncthreads();                             // tile ki ready in Wst[cur]
#else
    __syncthreads();
    stage_tile(&Wst[cur][0], W + (size_t)ki * 32 * Ndim + n0, 32, Ndim, tid);
    __syncthreads();
#endif
    v16bf a = load_a_bf16(arow, ki * 32 + hi * 8);
    const __bf16* wl = &Wst[cur][l15];
#pragma unroll
    for (int t = 0; t < 4; ++t)
      acc[t] = wmma_bf16(a, load_b_bf16(wl + t * 16), acc[t]);
    __syncthreads();                             // done reading before reuse
  }

#pragma unroll
  for (int t = 0; t < 4; ++t) {
    const int n   = n0 + t * 16 + l15;
    const float bv = bias[n];
#pragma unroll
    for (int r = 0; r < 8; ++r) {
      const int m = m0 + r + hi * 8;
      const float val = acc[t][r] + bv;
      if (mode == 0) {
        outF[(size_t)m * Ndim + n] = val;
      } else {
        const int bb = m >> 11;
        const int s  = m & (Sc - 1);
        const int h  = n >> 6;
        const int dk = n & (DKc - 1);
        const size_t addr = (mode == 1)
            ? ((size_t)((bb * Hc + h) * Sc + s) * DKc + dk)
            : ((size_t)((bb * Hc + h) * DKc + dk) * Sc + s);
        outB[addr] = (__bf16)val;
      }
    }
  }
}

// ---------------- flash attention ----------------
// 8 waves per block share one (b,h); consecutive 16-query tiles. K/V 64-key
// chunks are TDM-staged (padded) into LDS, double buffered, shared by all
// waves. Each wave: online softmax over its 16 rows, P staged in private LDS.
__global__ __launch_bounds__(256) void attn_kernel(
    const __bf16* __restrict__ Q,   // [B,H,S,DK]
    const __bf16* __restrict__ Kt,  // [B,H,DK,S]
    const __bf16* __restrict__ V,   // [B,H,S,DK]
    const int*    __restrict__ mask,// [S,S]
    __bf16*       __restrict__ outB)// [B,S,D] bf16 (feeds Wo GEMM)
{
  __shared__ __align__(16) __bf16 Kst[2][64 * LP];
  __shared__ __align__(16) __bf16 Vst[2][64 * LP];
  __shared__ __align__(16) __bf16 Pst[8][16 * LP];

  const int tid  = threadIdx.x;
  const int lane = tid & 31;
  const int wave = tid >> 5;
  const int l15  = lane & 15;
  const int hi   = lane >> 4;

  const int qTiles = Sc / 16;                    // 128 (divisible by 8)
  const int bh = blockIdx.x / (qTiles / 8);      // all 8 waves share (b,h)
  const int qt = (blockIdx.x % (qTiles / 8)) * 8 + wave;
  const int b  = bh / Hc;
  const int h  = bh % Hc;
  const int q0 = qt * 16;

  const __bf16* Qb  = Q  + (size_t)bh * Sc * DKc;
  const __bf16* Ktb = Kt + (size_t)bh * DKc * Sc;
  const __bf16* Vb  = V  + (size_t)bh * Sc * DKc;

  const __bf16* qrow = Qb + (size_t)(q0 + l15) * DKc;
  const v16bf aq0 = load_a_bf16(qrow, 0  + hi * 8);
  const v16bf aq1 = load_a_bf16(qrow, 32 + hi * 8);

  v8f   o[4] = {};
  float rmax[8], rsum[8];
#pragma unroll
  for (int r = 0; r < 8; ++r) { rmax[r] = -1e30f; rsum[r] = 0.f; }

  __bf16* P = &Pst[wave][0];
  constexpr int NCHUNK = Sc / 64;                // 32

#if HAVE_TDM
  if (wave == 0) {
    tdm_load_2d(&Kst[0][0], Ktb, Sc, DKc, 64, 64, Sc, true);
    tdm_load_2d(&Vst[0][0], Vb, DKc, Sc, 64, 64, DKc, true);
  }
#endif

  for (int kc = 0; kc < NCHUNK; ++kc) {
    const int cur = kc & 1;
    const int k0 = kc * 64;
#if HAVE_TDM
    if (wave == 0) {
      if (kc + 1 < NCHUNK) {
        const int kn = k0 + 64;
        tdm_load_2d(&Kst[cur ^ 1][0], Ktb + kn, Sc, DKc, 64, 64, Sc, true);
        tdm_load_2d(&Vst[cur ^ 1][0], Vb + (size_t)kn * DKc, DKc, Sc, 64, 64,
                    DKc, true);
        __builtin_amdgcn_s_wait_tensorcnt(2);
      } else {
        __builtin_amdgcn_s_wait_tensorcnt(0);
      }
    }
    __syncthreads();                             // chunk kc staged
#else
    __syncthreads();
    stage_tile(&Kst[cur][0], Ktb + k0, 64, Sc, tid);
    stage_tile(&Vst[cur][0], Vb + (size_t)k0 * DKc, 64, DKc, tid);
    __syncthreads();
#endif

    // -------- scores = Q * K^T (16x64 tile), K from LDS --------
    v8f s[4] = {};
    const __bf16* kl0 = &Kst[cur][l15];
    const __bf16* kl1 = &Kst[cur][32 * LP + l15];
#pragma unroll
    for (int t = 0; t < 4; ++t) {
      s[t] = wmma_bf16(aq0, load_b_bf16(kl0 + t * 16), s[t]);
      s[t] = wmma_bf16(aq1, load_b_bf16(kl1 + t * 16), s[t]);
    }

    // -------- scale, mask, per-row chunk max --------
    float cmax[8];
#pragma unroll
    for (int r = 0; r < 8; ++r) cmax[r] = -1e30f;
#pragma unroll
    for (int t = 0; t < 4; ++t) {
#pragma unroll
      for (int r = 0; r < 8; ++r) {
        float sv = s[t][r] * 0.125f;             // 1/sqrt(DK)
        const int mrow = q0 + r + hi * 8;
        const int mcol = k0 + t * 16 + l15;
        if (mask[(size_t)mrow * Sc + mcol] == 0) sv = -1e9f;
        s[t][r] = sv;
        cmax[r] = fmaxf(cmax[r], sv);
      }
    }
#pragma unroll
    for (int r = 0; r < 8; ++r) {
#pragma unroll
      for (int xm = 1; xm < 16; xm <<= 1)
        cmax[r] = fmaxf(cmax[r], __shfl_xor(cmax[r], xm, 32));
    }

    // -------- online softmax update --------
    float csum[8];
#pragma unroll
    for (int r = 0; r < 8; ++r) {
      const float nm   = fmaxf(rmax[r], cmax[r]);
      const float corr = __expf(rmax[r] - nm);
      rmax[r] = nm;
      rsum[r] = rsum[r] * corr;
#pragma unroll
      for (int t = 0; t < 4; ++t) o[t][r] = o[t][r] * corr;
      csum[r] = 0.f;
    }
#pragma unroll
    for (int t = 0; t < 4; ++t) {
#pragma unroll
      for (int r = 0; r < 8; ++r) {
        const float p = __expf(s[t][r] - rmax[r]);
        csum[r] += p;
        P[(r + hi * 8) * LP + t * 16 + l15] = (__bf16)p;  // C-layout -> rowmaj
      }
    }
#pragma unroll
    for (int r = 0; r < 8; ++r) {
#pragma unroll
      for (int xm = 1; xm < 16; xm <<= 1)
        csum[r] += __shfl_xor(csum[r], xm, 32);
      rsum[r] += csum[r];
    }

    // -------- O += P * V (both from LDS; LDS in-order per wave) --------
    const v16bf ap0 = load_a_bf16(P + (size_t)l15 * LP, 0  + hi * 8);
    const v16bf ap1 = load_a_bf16(P + (size_t)l15 * LP, 32 + hi * 8);
    const __bf16* vl0 = &Vst[cur][l15];
    const __bf16* vl1 = &Vst[cur][32 * LP + l15];
#pragma unroll
    for (int t = 0; t < 4; ++t) {
      o[t] = wmma_bf16(ap0, load_b_bf16(vl0 + t * 16), o[t]);
      o[t] = wmma_bf16(ap1, load_b_bf16(vl1 + t * 16), o[t]);
    }
    __syncthreads();                             // all reads done before reuse
  }

  // -------- normalize, write bf16 [B,S,D] for the Wo projection --------
#pragma unroll
  for (int t = 0; t < 4; ++t) {
#pragma unroll
    for (int r = 0; r < 8; ++r) {
      const int mrow = q0 + r + hi * 8;
      const int n    = h * DKc + t * 16 + l15;
      outB[((size_t)b * Sc + mrow) * Dc + n] = (__bf16)(o[t][r] / rsum[r]);
    }
  }
}

// ---------------- host launcher ----------------
extern "C" void kernel_launch(void* const* d_in, const int* in_sizes, int n_in,
                              void* d_out, int out_size, void* d_ws, size_t ws_size,
                              hipStream_t stream) {
  (void)in_sizes; (void)n_in; (void)out_size; (void)ws_size;

  const float* q    = (const float*)d_in[0];
  const float* k    = (const float*)d_in[1];
  const float* v    = (const float*)d_in[2];
  const int*   mask = (const int*)  d_in[3];
  const float* Wq   = (const float*)d_in[4];
  const float* bq   = (const float*)d_in[5];
  const float* Wk   = (const float*)d_in[6];
  const float* bk   = (const float*)d_in[7];
  const float* Wv   = (const float*)d_in[8];
  const float* bv   = (const float*)d_in[9];
  const float* Wo   = (const float*)d_in[10];
  const float* bo   = (const float*)d_in[11];

  // workspace layout (bf16 elements)
  const size_t nX = (size_t)Mtot * Dc;   // 4 Mi
  const size_t nW = (size_t)Dc * Dc;     // 1 Mi
  __bf16* Xq  = (__bf16*)d_ws;
  __bf16* Xk  = Xq  + nX;
  __bf16* Xv  = Xk  + nX;
  __bf16* Wqb = Xv  + nX;
  __bf16* Wkb = Wqb + nW;
  __bf16* Wvb = Wkb + nW;
  __bf16* Wob = Wvb + nW;
  __bf16* Qb  = Wob + nW;                // [B,H,S,DK]
  __bf16* Ktb = Qb  + nX;                // [B,H,DK,S]
  __bf16* Vb  = Ktb + nX;                // [B,H,S,DK]
  __bf16* Ab  = Vb  + nX;                // attention out [B,S,D]

  const int cvtXBlocks = (int)(nX / 8 / 256);    // 2048
  const int cvtWBlocks = (int)(nW / 8 / 256);    // 512
  cvt_kernel<<<cvtXBlocks, 256, 0, stream>>>(q,  Xq,  (int)nX);
  cvt_kernel<<<cvtXBlocks, 256, 0, stream>>>(k,  Xk,  (int)nX);
  cvt_kernel<<<cvtXBlocks, 256, 0, stream>>>(v,  Xv,  (int)nX);
  cvt_kernel<<<cvtWBlocks, 256, 0, stream>>>(Wq, Wqb, (int)nW);
  cvt_kernel<<<cvtWBlocks, 256, 0, stream>>>(Wk, Wkb, (int)nW);
  cvt_kernel<<<cvtWBlocks, 256, 0, stream>>>(Wv, Wvb, (int)nW);
  cvt_kernel<<<cvtWBlocks, 256, 0, stream>>>(Wo, Wob, (int)nW);

  const int gemmBlocks = (Mtot / 128) * (Ndim / 64);  // 512
  const int attnBlocks = Bc * Hc * (Sc / 16) / 8;     // 512

  proj_kernel<<<gemmBlocks, 256, 0, stream>>>(Xq, Wqb, bq, nullptr, Qb,  1);
  proj_kernel<<<gemmBlocks, 256, 0, stream>>>(Xk, Wkb, bk, nullptr, Ktb, 2);
  proj_kernel<<<gemmBlocks, 256, 0, stream>>>(Xv, Wvb, bv, nullptr, Vb,  1);
  attn_kernel<<<attnBlocks, 256, 0, stream>>>(Qb, Ktb, Vb, mask, Ab);
  proj_kernel<<<gemmBlocks, 256, 0, stream>>>(Ab, Wob, bo, (float*)d_out,
                                              nullptr, 0);
}
